// LocalAttentionLayer_11441792877189
// MI455X (gfx1250) — matrix operand order repaired and checked by reading, not compiled, not run on hardware
//
#include <hip/hip_runtime.h>

typedef unsigned short u16;
typedef unsigned int   u32;

typedef __attribute__((ext_vector_type(16))) __bf16 v16bf;
typedef __attribute__((ext_vector_type(8)))  float  v8f;
typedef int v4i_ __attribute__((vector_size(16)));   // int4 as the async builtin expects

union AFrag { v16bf v; u32 p[8]; };

#define S_LEN   2048
#define D_MODEL 2048
#define HD      128
#define NH      16
#define NKV     4
#define KVD     512     // NKV*HD
#define WIN     1024

#if defined(__has_builtin)
#if __has_builtin(__builtin_amdgcn_global_load_async_to_lds_b128)
#define HAVE_ASYNC 1
#endif
#endif

__device__ __forceinline__ u16 f2bf(float f) {
  u32 u = __float_as_uint(f);
  u32 r = (u + 0x7FFFu + ((u >> 16) & 1u)) >> 16;   // round-to-nearest-even
  return (u16)r;
}

// K index (even base of a bf16 pair) for A-matrix 16x32 fragment pair p,
// contiguous order (matches B): lanes 0-15 hold K {0..15}, lanes 16-31 {16..31}
__device__ __forceinline__ int a_kpair(int p, int grp) {
  return grp * 16 + 2 * p;
}

// 16-byte global->LDS copy: async (ASYNCcnt-tracked, no VGPR round trip) when available
__device__ __forceinline__ void cp16(const u16* g, u16* l) {
#ifdef HAVE_ASYNC
  __builtin_amdgcn_global_load_async_to_lds_b128(
      (__attribute__((address_space(1))) v4i_*)const_cast<u16*>(g),
      (__attribute__((address_space(3))) v4i_*)l, 0, 0);
#else
  *(uint4*)l = *(const uint4*)g;
#endif
}
__device__ __forceinline__ void cp_fence() {
#ifdef HAVE_ASYNC
#if __has_builtin(__builtin_amdgcn_s_wait_asynccnt)
  __builtin_amdgcn_s_wait_asynccnt(0);
#else
  asm volatile("s_wait_asynccnt 0" ::: "memory");
#endif
#endif
}

// ---------------------------------------------------------------------------
// f32 -> bf16 elementwise
// ---------------------------------------------------------------------------
__global__ void cvt_bf16_kernel(const float* __restrict__ in, u16* __restrict__ out, int n) {
  int i = blockIdx.x * blockDim.x + threadIdx.x;
  if (i < n) out[i] = f2bf(in[i]);
}

// ---------------------------------------------------------------------------
// f32 W[K][N] -> bf16 Wt[N][K]   (one-time weight transpose, LDS-tiled)
// block (32,8), grid (N/32, K/32)
// ---------------------------------------------------------------------------
__global__ __launch_bounds__(256) void transpose_bf16_kernel(
    const float* __restrict__ W, u16* __restrict__ Wt, int N, int K) {
  __shared__ float tile[32][33];
  int nb = blockIdx.x * 32, kb = blockIdx.y * 32;
  int tx = threadIdx.x, ty = threadIdx.y;
#pragma unroll
  for (int j = 0; j < 4; ++j)
    tile[ty + 8 * j][tx] = W[(size_t)(kb + ty + 8 * j) * N + nb + tx];
  __syncthreads();
#pragma unroll
  for (int j = 0; j < 4; ++j)
    Wt[(size_t)(nb + ty + 8 * j) * K + kb + tx] = f2bf(tile[tx][ty + 8 * j]);
}

// ---------------------------------------------------------------------------
// RoPE cos/sin table: tab[s*64 + i] = (cos, sin) of s * theta^(-2i/128)
// ---------------------------------------------------------------------------
__global__ void rope_tab_kernel(float2* __restrict__ tab) {
  int idx = blockIdx.x * blockDim.x + threadIdx.x;
  if (idx >= S_LEN * (HD / 2)) return;
  int s = idx >> 6, i = idx & 63;
  float invf = __expf(-(float)(2 * i) * (9.2103403719761836f / (float)HD));
  float ang = (float)s * invf;
  tab[idx] = make_float2(cosf(ang), sinf(ang));
}

// ---------------------------------------------------------------------------
// bf16 GEMM:  C[M,N] = A[M,2048] * Bt[N,2048]^T   (Bt pre-transposed [n][k])
// block tile 128(M) x 128(N), 256 threads = 8 waves; wave tile 16(M) x 128(N).
// Double-buffered async LDS staging: one fence+barrier per k-step.
// EP: 0 = f32 store, 1 = RoPE + bf16, 2 = bf16, 3 = bf16 transposed (b,feat,s)
// ---------------------------------------------------------------------------
template <int EP>
__global__ __launch_bounds__(256) void gemm_bf16_kernel(
    const u16* __restrict__ A, const u16* __restrict__ Bt,
    float* __restrict__ Cf, u16* __restrict__ Cbf,
    const float2* __restrict__ rope, int N) {

  __shared__ u16 sA[2][128][32];    // [buf][m][k]
  __shared__ u16 sBt[2][128][32];   // [buf][n][k]

  const int n0   = blockIdx.x * 128;
  const int m0   = blockIdx.y * 128;
  const int tid  = threadIdx.x;
  const int wid  = tid >> 5;        // wave = 16-row slice of M
  const int lane = tid & 31;
  const int grp  = lane >> 4;
  const int ln   = lane & 15;

  v8f acc[8];
#pragma unroll
  for (int c = 0; c < 8; ++c)
#pragma unroll
    for (int e = 0; e < 8; ++e) acc[c][e] = 0.f;

  const int sr = tid >> 1, sc2 = (tid & 1) * 16;   // stage: 128 rows x 32 cols, 16 u16/thread

  auto stage = [&](int buf, int kt) {
    const u16* ga = A + (size_t)(m0 + sr) * D_MODEL + kt + sc2;
    cp16(ga,     &sA[buf][sr][sc2]);
    cp16(ga + 8, &sA[buf][sr][sc2 + 8]);
    const u16* gb = Bt + (size_t)(n0 + sr) * D_MODEL + kt + sc2;
    cp16(gb,     &sBt[buf][sr][sc2]);
    cp16(gb + 8, &sBt[buf][sr][sc2 + 8]);
  };

  stage(0, 0);
  cp_fence();
  __syncthreads();

  for (int kt = 0; kt < D_MODEL; kt += 32) {
    const int cur = (kt >> 5) & 1;
    if (kt + 32 < D_MODEL) {
      stage(cur ^ 1, kt + 32);            // async loads overlap the WMMAs below
      if (kt + 64 < D_MODEL) {            // prefetch one more tile ahead
        __builtin_prefetch(A  + (size_t)(m0 + sr) * D_MODEL + kt + 64 + sc2, 0, 1);
        __builtin_prefetch(Bt + (size_t)(n0 + sr) * D_MODEL + kt + 64 + sc2, 0, 1);
      }
    }

    AFrag a;
#pragma unroll
    for (int p = 0; p < 8; ++p)
      a.p[p] = *(const u32*)&sA[cur][wid * 16 + ln][a_kpair(p, grp)];

#pragma unroll
    for (int c = 0; c < 8; ++c) {
      AFrag bt;
#pragma unroll
      for (int v = 0; v < 8; ++v)
        bt.p[v] = *(const u32*)&sBt[cur][c * 16 + ln][grp * 16 + 2 * v];
      acc[c] = __builtin_amdgcn_wmma_f32_16x16x32_bf16(
          false, a.v, false, bt.v, (short)0, acc[c], false, false);
    }

    cp_fence();        // next tile resident
    __syncthreads();   // all waves done reading 'cur' before it is rewritten
  }

#pragma unroll
  for (int c = 0; c < 8; ++c) {
#pragma unroll
    for (int r = 0; r < 8; ++r) {
      int row = m0 + wid * 16 + r + 8 * grp;
      int col = n0 + c * 16 + ln;
      float val = acc[c][r];
      if (EP == 0) {
        Cf[(size_t)row * N + col] = val;
      } else if (EP == 2) {
        Cbf[(size_t)row * N + col] = f2bf(val);
      } else if (EP == 3) {
        // V: store transposed (b, feature, s) for the attention PV B-operand
        Cbf[((size_t)(row >> 11) * N + col) * S_LEN + (row & (S_LEN - 1))] = f2bf(val);
      } else {
        int s  = row & (S_LEN - 1);
        int i2 = (col & (HD - 1)) >> 1;
        float2 cs2 = rope[(s << 6) + i2];
        float other = __shfl_xor(val, 1, 32);
        bool evenc = ((col & 1) == 0);
        float x0 = evenc ? val : other;
        float x1 = evenc ? other : val;
        float res = evenc ? (x0 * cs2.x - x1 * cs2.y) : (x0 * cs2.y + x1 * cs2.x);
        Cbf[(size_t)row * N + col] = f2bf(res);
      }
    }
  }
}

// ---------------------------------------------------------------------------
// Sliding-window GQA flash attention, double-buffered K/V staging.
// grid (S/64, NH, B); block 128 = 4 waves; wave owns 16 q rows.
// Q (b,s,h,d) stride 2048; K (b,s,kvh,d) stride 512; Vt (b, kvh*128+d, s).
// ---------------------------------------------------------------------------
__global__ __launch_bounds__(128) void attn_kernel(
    const u16* __restrict__ Q, const u16* __restrict__ Kd,
    const u16* __restrict__ Vt, u16* __restrict__ O) {

  __shared__ u16 sQ[64][HD];          // [q][d]
  __shared__ u16 sK[2][32][HD];       // [buf][key][d]
  __shared__ u16 sVt[2][HD][32];      // [buf][d][key]
  __shared__ u16 sP[4][16][32];       // per-wave probabilities

  const int qb  = blockIdx.x;
  const int h   = blockIdx.y;
  const int b   = blockIdx.z;
  const int kvh = h >> 2;
  const int i0  = qb * 64;

  const int tid  = threadIdx.x;
  const int wid  = tid >> 5;
  const int lane = tid & 31;
  const int grp  = lane >> 4;
  const int ln   = lane & 15;

  const int kr = tid >> 2, kc = (tid & 3) * 32;   // K-stage mapping
  const u16* vrow = Vt + ((size_t)(b * KVD + kvh * HD + tid)) * S_LEN;

  auto stageKV = [&](int buf, int jt) {
    const u16* g = Kd + ((size_t)(b * S_LEN + jt + kr)) * (size_t)KVD + kvh * HD + kc;
#pragma unroll
    for (int u = 0; u < 4; ++u) cp16(g + u * 8, &sK[buf][kr][kc + u * 8]);
#pragma unroll
    for (int u = 0; u < 4; ++u) cp16(vrow + jt + u * 8, &sVt[buf][tid][u * 8]);
  };

  int jlo = i0 - WIN; if (jlo < 0) jlo = 0;

  {   // stage Q tile 64x128 (and first K/V tile) up front
    int r = tid >> 1, c0 = (tid & 1) * 64;
    const u16* g = Q + ((size_t)(b * S_LEN + i0 + r)) * (size_t)D_MODEL + h * HD + c0;
#pragma unroll
    for (int u = 0; u < 8; ++u) cp16(g + u * 8, &sQ[r][c0 + u * 8]);
  }
  stageKV(0, jlo);
  cp_fence();
  __syncthreads();

  AFrag aq[4];     // Q fragments live in registers for the whole key loop
#pragma unroll
  for (int ks = 0; ks < 4; ++ks)
#pragma unroll
    for (int p = 0; p < 8; ++p)
      aq[ks].p[p] = *(const u32*)&sQ[wid * 16 + ln][ks * 32 + a_kpair(p, grp)];

  float m_r[8], l_r[8];
  v8f o_acc[8];
#pragma unroll
  for (int r = 0; r < 8; ++r) { m_r[r] = -1e30f; l_r[r] = 0.f; }
#pragma unroll
  for (int c = 0; c < 8; ++c)
#pragma unroll
    for (int e = 0; e < 8; ++e) o_acc[c][e] = 0.f;

  const float scale = 0.08838834764831845f;   // 1/sqrt(128)

  for (int jt = jlo; jt < i0 + 64; jt += 32) {
    const int cur = ((jt - jlo) >> 5) & 1;
    if (jt + 32 < i0 + 64) {
      stageKV(cur ^ 1, jt + 32);      // async loads overlap scores+softmax+PV
      __builtin_prefetch(Kd + ((size_t)(b * S_LEN + jt + 64 + kr)) * (size_t)KVD + kvh * HD + kc, 0, 1);
      __builtin_prefetch(vrow + jt + 64, 0, 1);
    }

    // scores: 16 q rows x 32 keys, chained over d (K=128)
    v8f sc[2];
#pragma unroll
    for (int ch = 0; ch < 2; ++ch) {
      v8f a;
#pragma unroll
      for (int e = 0; e < 8; ++e) a[e] = 0.f;
#pragma unroll
      for (int ks = 0; ks < 4; ++ks) {
        AFrag bk;   // B = K^T: element = sK[key][d], pairs contiguous in d
#pragma unroll
        for (int v = 0; v < 8; ++v)
          bk.p[v] = *(const u32*)&sK[cur][ch * 16 + ln][ks * 32 + grp * 16 + 2 * v];
        a = __builtin_amdgcn_wmma_f32_16x16x32_bf16(
            false, aq[ks].v, false, bk.v, (short)0, a, false, false);
      }
      sc[ch] = a;
    }

    // banded causal mask: keep j <= i && j >= i - WIN
#pragma unroll
    for (int ch = 0; ch < 2; ++ch) {
      int j = jt + ch * 16 + ln;
#pragma unroll
      for (int r = 0; r < 8; ++r) {
        int i = i0 + wid * 16 + r + 8 * grp;
        float s = sc[ch][r] * scale;
        bool ok = (j <= i) && (j + WIN >= i);
        sc[ch][r] = ok ? s : -1e30f;
      }
    }

    // online softmax (row lives in one 16-lane half -> shfl_xor 1..8)
#pragma unroll
    for (int r = 0; r < 8; ++r) {
      float mx = fmaxf(sc[0][r], sc[1][r]);
#pragma unroll
      for (int d = 1; d < 16; d <<= 1) mx = fmaxf(mx, __shfl_xor(mx, d, 32));
      float mnew  = fmaxf(m_r[r], mx);
      float alpha = __expf(m_r[r] - mnew);
      float p0 = (sc[0][r] > -9e29f) ? __expf(sc[0][r] - mnew) : 0.f;
      float p1 = (sc[1][r] > -9e29f) ? __expf(sc[1][r] - mnew) : 0.f;
      float rs = p0 + p1;
#pragma unroll
      for (int d = 1; d < 16; d <<= 1) rs += __shfl_xor(rs, d, 32);
      l_r[r] = l_r[r] * alpha + rs;
      m_r[r] = mnew;
#pragma unroll
      for (int c = 0; c < 8; ++c) o_acc[c][r] *= alpha;
      sP[wid][r + 8 * grp][ln]      = f2bf(p0);
      sP[wid][r + 8 * grp][16 + ln] = f2bf(p1);
    }
    __syncthreads();   // sP cross-lane visibility

    // O += P(16x32) @ V(32x128)
    AFrag ap;
#pragma unroll
    for (int p = 0; p < 8; ++p)
      ap.p[p] = *(const u32*)&sP[wid][ln][a_kpair(p, grp)];
#pragma unroll
    for (int c = 0; c < 8; ++c) {
      AFrag bv;   // element = sVt[d][key], pairs contiguous in key
#pragma unroll
      for (int v = 0; v < 8; ++v)
        bv.p[v] = *(const u32*)&sVt[cur][c * 16 + ln][grp * 16 + 2 * v];
      o_acc[c] = __builtin_amdgcn_wmma_f32_16x16x32_bf16(
          false, ap.v, false, bv.v, (short)0, o_acc[c], false, false);
    }

    cp_fence();        // next K/V tile resident
    __syncthreads();   // all waves done with 'cur' before rewrite
  }

#pragma unroll
  for (int r = 0; r < 8; ++r) {
    float inv = 1.f / l_r[r];
    int srow = i0 + wid * 16 + r + 8 * grp;
#pragma unroll
    for (int c = 0; c < 8; ++c) {
      O[((size_t)(b * S_LEN + srow)) * (size_t)D_MODEL + h * HD + c * 16 + ln] =
          f2bf(o_acc[c][r] * inv);
    }
  }
}

// ---------------------------------------------------------------------------
extern "C" void kernel_launch(void* const* d_in, const int* in_sizes, int n_in,
                              void* d_out, int out_size, void* d_ws, size_t ws_size,
                              hipStream_t stream) {
  const float* x  = (const float*)d_in[0];
  const float* wq = (const float*)d_in[1];
  const float* wk = (const float*)d_in[2];
  const float* wv = (const float*)d_in[3];
  const float* wo = (const float*)d_in[4];

  const int D = D_MODEL, S = S_LEN;
  const size_t M = (size_t)2 * S;   // 4096 rows

  u16* Xb  = (u16*)d_ws;                       // [M][D]
  u16* Wqt = Xb  + M * D;                      // [D][D]   transposed
  u16* Wkt = Wqt + (size_t)D * D;              // [512][D] transposed
  u16* Wvt = Wkt + (size_t)KVD * D;
  u16* Wot = Wvt + (size_t)KVD * D;            // [D][D]   transposed
  u16* Qr  = Wot + (size_t)D * D;              // (b,s,h,d)
  u16* Kr  = Qr  + M * D;                      // (b,s,kvh,d)
  u16* Vt  = Kr  + M * KVD;                    // (b, feature, s)
  u16* At  = Vt  + M * KVD;                    // (b,s,h,d)
  float2* ropeT = (float2*)(At + M * D);       // [S][64]

  cvt_bf16_kernel<<<(int)((M * D + 255) / 256), 256, 0, stream>>>(x, Xb, (int)(M * D));
  transpose_bf16_kernel<<<dim3(D / 32,   D / 32), dim3(32, 8), 0, stream>>>(wq, Wqt, D,   D);
  transpose_bf16_kernel<<<dim3(KVD / 32, D / 32), dim3(32, 8), 0, stream>>>(wk, Wkt, KVD, D);
  transpose_bf16_kernel<<<dim3(KVD / 32, D / 32), dim3(32, 8), 0, stream>>>(wv, Wvt, KVD, D);
  transpose_bf16_kernel<<<dim3(D / 32,   D / 32), dim3(32, 8), 0, stream>>>(wo, Wot, D,   D);
  rope_tab_kernel<<<(S * (HD / 2) + 255) / 256, 256, 0, stream>>>(ropeT);

  dim3 blk(256);
  gemm_bf16_kernel<1><<<dim3(D / 128,   (int)(M / 128)), blk, 0, stream>>>(Xb, Wqt, nullptr, Qr, ropeT, D);
  gemm_bf16_kernel<1><<<dim3(KVD / 128, (int)(M / 128)), blk, 0, stream>>>(Xb, Wkt, nullptr, Kr, ropeT, KVD);
  gemm_bf16_kernel<3><<<dim3(KVD / 128, (int)(M / 128)), blk, 0, stream>>>(Xb, Wvt, nullptr, Vt, ropeT, KVD);

  attn_kernel<<<dim3(S / 64, NH, 2), 128, 0, stream>>>(Qr, Kr, Vt, At);

  gemm_bf16_kernel<0><<<dim3(D / 128, (int)(M / 128)), blk, 0, stream>>>(At, Wot, (float*)d_out, nullptr, ropeT, D);
}
